// CausalSelfAttention_50663434224313
// MI455X (gfx1250) — matrix operand order repaired and checked by reading, not compiled, not run on hardware
//
#include <hip/hip_runtime.h>
#include <hip/hip_bf16.h>
#include <stdint.h>

// ---------------------------------------------------------------------------
// CDNA5 (gfx1250) wave32 WMMA attention pipeline
//   x(2,2048,1024) f32 -> ternary QKV gemm -> rmsnorm+rope -> causal GQA
//   flash attention -> rmsnorm -> ternary proj gemm -> out f32
// All matrix math through v_wmma_f32_16x16x32_bf16 (fp32 accumulate).
// ---------------------------------------------------------------------------

typedef __attribute__((ext_vector_type(16))) __bf16 v16bf;
typedef __attribute__((ext_vector_type(8)))  float  v8f;

union Frag16 { v16bf v; unsigned u[8]; unsigned short h[16]; };
union Q4u    { uint4 q; unsigned short h[8]; };

__device__ inline unsigned short f2bf(float f) {
  unsigned x = __float_as_uint(f);
  unsigned r = x + 0x7FFFu + ((x >> 16) & 1u);   // round-to-nearest-even
  return (unsigned short)(r >> 16);
}
__device__ inline float bf2f(unsigned short h) {
  return __uint_as_float(((unsigned)h) << 16);
}
__device__ inline v8f v8f_zero() { v8f z; for (int i = 0; i < 8; ++i) z[i] = 0.f; return z; }

__device__ inline v8f wmma_bf16(v16bf a, v16bf b, v8f c) {
  // (neg_a, A, neg_b, B, c_mod, C, reuse_a, reuse_b)
  return __builtin_amdgcn_wmma_f32_16x16x32_bf16(false, a, false, b, (short)0, c, false, false);
}

// Wave-internal LDS producer->consumer fence (all lanes of one wave).
__device__ inline void lds_fence_wave() {
  __builtin_amdgcn_wave_barrier();
  asm volatile("s_wait_dscnt 0x0" ::: "memory");
  __builtin_amdgcn_wave_barrier();
}

#define BATCH   2
#define SEQ     2048
#define DIM     1024
#define NHEADS  16
#define NKVH    4
#define HDIM    64
#define NTOK    (BATCH * SEQ)           // 4096
#define QKV_N   (DIM + 2 * NKVH * HDIM) // 1536

// ---------------------------------------------------------------------------
// 1) ternary_weight: per-64-group  scale = max(mean|bf16(w)|, 1e-8),
//    q = clip(round(w/scale), -1, 1), store bf16(q*scale)
// ---------------------------------------------------------------------------
__global__ void csa_quant_ternary(const float* __restrict__ W,
                                  unsigned short* __restrict__ Wq,
                                  int ngroups) {
  int g = blockIdx.x * blockDim.x + threadIdx.x;
  if (g >= ngroups) return;
  const float* wg = W + (size_t)g * 64;
  float wb[64];
  float s = 0.f;
  for (int i = 0; i < 64; ++i) {
    wb[i] = bf2f(f2bf(wg[i]));
    s += fabsf(wb[i]);
  }
  s *= (1.f / 64.f);
  s = fmaxf(s, 1e-8f);
  float invs = 1.f / s;
  for (int i = 0; i < 64; ++i) {
    float q = rintf(wb[i] * invs);
    q = fminf(fmaxf(q, -1.f), 1.f);
    Wq[(size_t)g * 64 + i] = f2bf(q * s);
  }
}

// ---------------------------------------------------------------------------
// 2) cast fp32 activations -> bf16
// ---------------------------------------------------------------------------
__global__ void csa_cast_bf16(const float* __restrict__ X,
                              unsigned short* __restrict__ Xb, int n) {
  int i = blockIdx.x * blockDim.x + threadIdx.x;
  if (i < n) Xb[i] = f2bf(X[i]);
}

// ---------------------------------------------------------------------------
// 3) Generic bf16 WMMA GEMM:  C[M,N] f32 = A[M,K] * B[N,K]^T
//    256 threads = 8 waves; block tile 128x128; K staged 64-wide in LDS.
//    Wave (wm,wn): 32x64 tile -> 2x4 accumulators of 16x16.
//    Staging: batch all global b128 loads into registers, single wait,
//    then LDS stores (avoids load->wait->store serialization).
// ---------------------------------------------------------------------------
__global__ __launch_bounds__(256) void csa_gemm_bf16_wmma(
    const unsigned short* __restrict__ A,
    const unsigned short* __restrict__ B,
    float* __restrict__ C, int M, int N, int K) {
  __shared__ unsigned short As[128][72];   // +8 pad to dodge bank conflicts
  __shared__ unsigned short Bs[128][72];

  const int tid  = threadIdx.x;
  const int lane = tid & 31;
  const int wave = tid >> 5;
  const int wm = (wave >> 1) * 32;   // 0,32,64,96
  const int wn = (wave & 1) * 64;    // 0,64
  const int gm0 = blockIdx.y * 128;
  const int gn0 = blockIdx.x * 128;
  const int half = lane >> 4;        // 0/1
  const int l15  = lane & 15;

  v8f acc[2][4];
  for (int mi = 0; mi < 2; ++mi)
    for (int ni = 0; ni < 4; ++ni) acc[mi][ni] = v8f_zero();

  for (int kc = 0; kc < K; kc += 64) {
    // cooperative stage: 128 rows x 64 cols of each operand
    uint4 ra[4], rb[4];
    for (int i = 0; i < 4; ++i) {
      int idx = tid + i * 256;            // 0..1023
      int row = idx >> 3;
      int c8  = (idx & 7) * 8;
      ra[i] = *(const uint4*)&A[(size_t)(gm0 + row) * K + kc + c8];
      rb[i] = *(const uint4*)&B[(size_t)(gn0 + row) * K + kc + c8];
    }
    if (kc + 64 < K) {
      __builtin_prefetch(&A[(size_t)(gm0 + (tid >> 1)) * K + kc + 64], 0, 1);
      __builtin_prefetch(&B[(size_t)(gn0 + (tid >> 1)) * K + kc + 64], 0, 1);
    }
    for (int i = 0; i < 4; ++i) {
      int idx = tid + i * 256;
      int row = idx >> 3;
      int c8  = (idx & 7) * 8;
      *(uint4*)&As[row][c8] = ra[i];
      *(uint4*)&Bs[row][c8] = rb[i];
    }
    __syncthreads();

    for (int ks = 0; ks < 64; ks += 32) {
      Frag16 a[2], b[4];
      // A fragment (16x32 bf16): lane m = lane&15; VGPR i holds K pair
      for (int mi = 0; mi < 2; ++mi) {
        int r = wm + mi * 16 + l15;
        for (int i = 0; i < 8; ++i) {
          int kk = (i < 4) ? (half * 8 + i * 2)
                           : (16 + half * 8 + (i - 4) * 2);
          a[mi].u[i] = *(const unsigned*)&As[r][ks + kk];
        }
      }
      // B fragment (32x16 bf16): lane n = lane&15; VGPR i holds K pair (same order as A)
      for (int ni = 0; ni < 4; ++ni) {
        int cn = wn + ni * 16 + l15;
        for (int i = 0; i < 8; ++i) {
          int kk = (i < 4) ? (half * 8 + i * 2) : (16 + half * 8 + (i - 4) * 2);
          b[ni].u[i] = *(const unsigned*)&Bs[cn][ks + kk];
        }
      }
      for (int mi = 0; mi < 2; ++mi)
        for (int ni = 0; ni < 4; ++ni)
          acc[mi][ni] = wmma_bf16(a[mi].v, b[ni].v, acc[mi][ni]);
    }
    __syncthreads();
  }

  // C/D layout: VGPR r -> M = half*8 + r, N = lane&15
  for (int mi = 0; mi < 2; ++mi)
    for (int ni = 0; ni < 4; ++ni)
      for (int r = 0; r < 8; ++r) {
        int row = gm0 + wm + mi * 16 + half * 8 + r;
        int col = gn0 + wn + ni * 16 + l15;
        C[(size_t)row * N + col] = acc[mi][ni][r];
      }
}

// ---------------------------------------------------------------------------
// 4) QKV epilogue: per-head rmsnorm + rope (+gain for q), emit bf16
//    q: [b][16][s][64]   k,v: [b][4][s][64]
// ---------------------------------------------------------------------------
__global__ void csa_qkv_post(const float* __restrict__ qkv,
                             const float* __restrict__ qgain,
                             unsigned short* __restrict__ Qo,
                             unsigned short* __restrict__ Ko,
                             unsigned short* __restrict__ Vo) {
  int idx = blockIdx.x * blockDim.x + threadIdx.x;
  if (idx >= NTOK * 24) return;
  int token = idx / 24, slot = idx % 24;
  int b = token >> 11, s = token & (SEQ - 1);

  float vals[HDIM];
  const float eps = 1.1920929e-7f;

  if (slot < 16) {              // q head
    int h = slot;
    const float* src = qkv + (size_t)token * QKV_N + h * HDIM;
    float ss = 0.f;
    for (int i = 0; i < HDIM; ++i) { vals[i] = src[i]; ss += vals[i] * vals[i]; }
    float inv = rsqrtf(ss * (1.f / HDIM) + eps);
    float g = qgain[h];
    unsigned short* dst = Qo + (((size_t)b * NHEADS + h) * SEQ + s) * HDIM;
    for (int i = 0; i < 32; ++i) {
      float x1 = vals[i] * inv, x2 = vals[i + 32] * inv;
      float ang = (float)s * __powf(10000.f, -(float)i / 32.f);
      float c = __cosf(ang), sn = __sinf(ang);
      dst[i]      = f2bf((x1 * c + x2 * sn) * g);
      dst[i + 32] = f2bf((-x1 * sn + x2 * c) * g);
    }
  } else if (slot < 20) {       // k head
    int h = slot - 16;
    const float* src = qkv + (size_t)token * QKV_N + DIM + h * HDIM;
    float ss = 0.f;
    for (int i = 0; i < HDIM; ++i) { vals[i] = src[i]; ss += vals[i] * vals[i]; }
    float inv = rsqrtf(ss * (1.f / HDIM) + eps);
    unsigned short* dst = Ko + (((size_t)b * NKVH + h) * SEQ + s) * HDIM;
    for (int i = 0; i < 32; ++i) {
      float x1 = vals[i] * inv, x2 = vals[i + 32] * inv;
      float ang = (float)s * __powf(10000.f, -(float)i / 32.f);
      float c = __cosf(ang), sn = __sinf(ang);
      dst[i]      = f2bf(x1 * c + x2 * sn);
      dst[i + 32] = f2bf(-x1 * sn + x2 * c);
    }
  } else {                      // v head
    int h = slot - 20;
    const float* src = qkv + (size_t)token * QKV_N + DIM + NKVH * HDIM + h * HDIM;
    unsigned short* dst = Vo + (((size_t)b * NKVH + h) * SEQ + s) * HDIM;
    for (int i = 0; i < HDIM; ++i) dst[i] = f2bf(src[i]);
  }
}

// ---------------------------------------------------------------------------
// 5) Flash attention: one wave per 16-query tile, 32-key blocks.
//    K/V tiles staged per-wave in LDS with coalesced b128 global loads
//    (V transposed to [d][key] so PV B-fragments are contiguous b32 reads).
//    S = Q K^T, online softmax, P->A-frag via per-wave LDS transpose,
//    PV via 4 WMMA per block.
// ---------------------------------------------------------------------------
__global__ __launch_bounds__(128) void csa_attn_wmma(
    const unsigned short* __restrict__ Qm,
    const unsigned short* __restrict__ Km,
    const unsigned short* __restrict__ Vm,
    float* __restrict__ Y) {
  __shared__ unsigned short Ks[4][32][72];   // [wave][key][d]  (+8 pad)
  __shared__ unsigned short Vt[4][64][36];   // [wave][d][key]  (+4 pad)
  __shared__ unsigned short Pl[4][16][40];   // per-wave P tile (16x32 + pad)

  const int lane = threadIdx.x & 31;
  const int wave = threadIdx.x >> 5;
  const int it = blockIdx.x * 4 + wave;      // q tile index (0..127)
  const int h = blockIdx.y, b = blockIdx.z;
  const int kvh = h >> 2;                    // GQA: 16 heads -> 4 kv heads
  const int q0 = it * 16;
  const int half = lane >> 4, l15 = lane & 15;
  const float scale = 0.125f;                // 1/sqrt(64)

  const size_t qbase = (((size_t)b * NHEADS + h) * SEQ) * HDIM;
  const size_t kbase = (((size_t)b * NKVH + kvh) * SEQ) * HDIM;

  // Q fragments, d split into 2 chunks of 32
  Frag16 qa[2];
  for (int c = 0; c < 2; ++c)
    for (int i = 0; i < 8; ++i) {
      int kk = (i < 4) ? (half * 8 + i * 2) : (16 + half * 8 + (i - 4) * 2);
      qa[c].u[i] =
          *(const unsigned*)&Qm[qbase + (size_t)(q0 + l15) * HDIM + c * 32 + kk];
    }

  v8f acc[4];
  for (int c = 0; c < 4; ++c) acc[c] = v8f_zero();
  float mrow[8], lrow[8];
  for (int r = 0; r < 8; ++r) { mrow[r] = -1e30f; lrow[r] = 0.f; }

  const int jmax = (q0 + 16 + 31) >> 5;      // 32-key blocks (causal bound)
  for (int jb = 0; jb < jmax; ++jb) {
    const int keyb = jb * 32;

    // ---- stage K tile (as-is) and V tile (transposed) into per-wave LDS ----
    {
      Q4u r[8];
      for (int i = 0; i < 8; ++i) {          // K: 32 rows x 64 d = 256 vec8
        int seg = i * 32 + lane;
        int row = seg >> 3, c8 = (seg & 7) * 8;
        r[i].q = *(const uint4*)&Km[kbase + (size_t)(keyb + row) * HDIM + c8];
      }
      for (int i = 0; i < 8; ++i) {
        int seg = i * 32 + lane;
        int row = seg >> 3, c8 = (seg & 7) * 8;
        *(uint4*)&Ks[wave][row][c8] = r[i].q;
      }
      for (int i = 0; i < 8; ++i) {          // V: same tile, scatter-transpose
        int seg = i * 32 + lane;
        int row = seg >> 3, c8 = (seg & 7) * 8;
        r[i].q = *(const uint4*)&Vm[kbase + (size_t)(keyb + row) * HDIM + c8];
      }
      for (int i = 0; i < 8; ++i) {
        int seg = i * 32 + lane;
        int row = seg >> 3, c8 = (seg & 7) * 8;
        for (int e = 0; e < 8; ++e) Vt[wave][c8 + e][row] = r[i].h[e];
      }
    }
    lds_fence_wave();

    // ---- S = Q K^T ----
    v8f S0 = v8f_zero(), S1 = v8f_zero();
    for (int c = 0; c < 2; ++c) {
      Frag16 bk0, bk1;
      for (int i = 0; i < 8; ++i) {
        int kk = (i < 4) ? (half * 8 + i * 2) : (16 + half * 8 + (i - 4) * 2);          // d index within chunk
        bk0.u[i] = *(const unsigned*)&Ks[wave][l15][c * 32 + kk];
        bk1.u[i] = *(const unsigned*)&Ks[wave][16 + l15][c * 32 + kk];
      }
      S0 = wmma_bf16(qa[c].v, bk0.v, S0);
      S1 = wmma_bf16(qa[c].v, bk1.v, S1);
    }

    // causal mask + scale + online softmax (row = half*8 + r across 16 lanes)
    float mnew[8], p0[8], p1[8];
    for (int r = 0; r < 8; ++r) {
      int row = q0 + half * 8 + r;
      float s0 = (keyb + l15 <= row) ? S0[r] * scale : -1e30f;
      float s1 = (keyb + 16 + l15 <= row) ? S1[r] * scale : -1e30f;
      float mx = fmaxf(s0, s1);
      mx = fmaxf(mx, __shfl_xor(mx, 1, 32));
      mx = fmaxf(mx, __shfl_xor(mx, 2, 32));
      mx = fmaxf(mx, __shfl_xor(mx, 4, 32));
      mx = fmaxf(mx, __shfl_xor(mx, 8, 32));
      mnew[r] = fmaxf(mrow[r], mx);
      float alpha = __expf(mrow[r] - mnew[r]);
      p0[r] = __expf(s0 - mnew[r]);
      p1[r] = __expf(s1 - mnew[r]);
      float rs = p0[r] + p1[r];
      rs += __shfl_xor(rs, 1, 32);
      rs += __shfl_xor(rs, 2, 32);
      rs += __shfl_xor(rs, 4, 32);
      rs += __shfl_xor(rs, 8, 32);
      lrow[r] = lrow[r] * alpha + rs;
      mrow[r] = mnew[r];
      for (int c = 0; c < 4; ++c) acc[c][r] *= alpha;
      Pl[wave][half * 8 + r][l15]      = f2bf(p0[r]);
      Pl[wave][half * 8 + r][16 + l15] = f2bf(p1[r]);
    }
    lds_fence_wave();

    // P as A fragment (16 rows x 32 keys)
    Frag16 pa;
    for (int i = 0; i < 8; ++i) {
      int kk = (i < 4) ? (half * 8 + i * 2) : (16 + half * 8 + (i - 4) * 2);
      pa.u[i] = *(const unsigned*)&Pl[wave][l15][kk];
    }
    lds_fence_wave();

    // ---- PV: B-frag per 16-wide d chunk, contiguous b32 reads from Vt ----
    for (int c = 0; c < 4; ++c) {
      Frag16 bv;
      for (int i = 0; i < 8; ++i) {
        int kk = (i < 4) ? (half * 8 + i * 2) : (16 + half * 8 + (i - 4) * 2);          // key pair index
        bv.u[i] = *(const unsigned*)&Vt[wave][c * 16 + l15][kk];
      }
      acc[c] = wmma_bf16(pa.v, bv.v, acc[c]);
    }
  }

  // normalize and write y as [b][s][h*64+d] fp32
  for (int r = 0; r < 8; ++r) {
    float inv = 1.f / lrow[r];
    int row = q0 + half * 8 + r;
    for (int c = 0; c < 4; ++c)
      Y[((size_t)(b * SEQ + row)) * DIM + h * HDIM + c * 16 + l15] =
          acc[c][r] * inv;
  }
}

// ---------------------------------------------------------------------------
// 6) rmsnorm over dim=1024 per token, emit bf16
// ---------------------------------------------------------------------------
__global__ __launch_bounds__(256) void csa_rmsnorm_rows(
    const float* __restrict__ Y, unsigned short* __restrict__ Yn, int D) {
  __shared__ float red[256];
  const int row = blockIdx.x;
  const float* yr = Y + (size_t)row * D;
  float ss = 0.f;
  for (int i = threadIdx.x; i < D; i += 256) { float t = yr[i]; ss += t * t; }
  red[threadIdx.x] = ss;
  __syncthreads();
  for (int st = 128; st > 0; st >>= 1) {
    if (threadIdx.x < st) red[threadIdx.x] += red[threadIdx.x + st];
    __syncthreads();
  }
  float inv = rsqrtf(red[0] / (float)D + 1.1920929e-7f);
  for (int i = threadIdx.x; i < D; i += 256)
    Yn[(size_t)row * D + i] = f2bf(yr[i] * inv);
}

// ---------------------------------------------------------------------------
// launch
// ---------------------------------------------------------------------------
extern "C" void kernel_launch(void* const* d_in, const int* in_sizes, int n_in,
                              void* d_out, int out_size, void* d_ws, size_t ws_size,
                              hipStream_t stream) {
  const float* x      = (const float*)d_in[0];   // (2,2048,1024)
  const float* w_qkv  = (const float*)d_in[1];   // (1536,1024)
  const float* w_proj = (const float*)d_in[2];   // (1024,1024)
  const float* q_gain = (const float*)d_in[3];   // (16,)
  float* out = (float*)d_out;
  char* ws = (char*)d_ws;

  // workspace layout (bytes); y aliases qkv, yn aliases xb after they die
  unsigned short* wqkv_b  = (unsigned short*)(ws + 0);          // 3,145,728
  unsigned short* wproj_b = (unsigned short*)(ws + 3145728);    // 2,097,152
  unsigned short* xb      = (unsigned short*)(ws + 5242880);    // 8,388,608
  float*          qkv     = (float*)(ws + 13631488);            // 25,165,824
  unsigned short* qb      = (unsigned short*)(ws + 38797312);   // 8,388,608
  unsigned short* kb      = (unsigned short*)(ws + 47185920);   // 2,097,152
  unsigned short* vb      = (unsigned short*)(ws + 49283072);   // 2,097,152
  float*          y       = (float*)(ws + 13631488);            // alias qkv
  unsigned short* yn      = (unsigned short*)(ws + 5242880);    // alias xb

  // 1) quantize weights
  csa_quant_ternary<<<(QKV_N * DIM / 64 + 255) / 256, 256, 0, stream>>>(
      w_qkv, wqkv_b, QKV_N * DIM / 64);
  csa_quant_ternary<<<(DIM * DIM / 64 + 255) / 256, 256, 0, stream>>>(
      w_proj, wproj_b, DIM * DIM / 64);

  // 2) cast x
  csa_cast_bf16<<<(NTOK * DIM + 255) / 256, 256, 0, stream>>>(x, xb, NTOK * DIM);

  // 3) qkv gemm: (4096 x 1024) * (1536 x 1024)^T
  csa_gemm_bf16_wmma<<<dim3(QKV_N / 128, NTOK / 128), 256, 0, stream>>>(
      xb, wqkv_b, qkv, NTOK, QKV_N, DIM);

  // 4) rmsnorm + rope + gain
  csa_qkv_post<<<(NTOK * 24 + 255) / 256, 256, 0, stream>>>(
      qkv, q_gain, qb, kb, vb);

  // 5) flash attention (y aliases qkv storage)
  csa_attn_wmma<<<dim3(SEQ / 16 / 4, NHEADS, BATCH), 128, 0, stream>>>(
      qb, kb, vb, y);

  // 6) rmsnorm rows (yn aliases xb storage)
  csa_rmsnorm_rows<<<NTOK, 256, 0, stream>>>(y, yn, DIM);

  // 7) proj gemm -> out
  csa_gemm_bf16_wmma<<<dim3(DIM / 128, NTOK / 128), 256, 0, stream>>>(
      yn, wproj_b, out, NTOK, DIM, DIM);
}
